// LSTM_Decoder_21268678050457
// MI455X (gfx1250) — matrix operand order repaired and checked by reading, not compiled, not run on hardware
//
#include <hip/hip_runtime.h>
#include <hip/hip_bf16.h>

// Problem dims
#define B_     16
#define T_     256
#define V_     32000
#define E_     512
#define H_     512
#define EH_    1024
#define G4H_   2048
#define STEPS_ 255

// Launch shape for the fused step kernel
#define GRID_  128     // blocks (must all be co-resident: grid barrier)
#define TPB_   256     // 8 waves per block (wave32)
#define NWAVES_ (GRID_ * 8)        // 1024 waves
#define PROJ_TILES_ (V_ / 16)      // 2000 column tiles of 16

typedef __attribute__((ext_vector_type(16))) __bf16 v16bf;
typedef __attribute__((ext_vector_type(8)))  float  v8f;

union BF16x16 { v16bf v; uint4 u[2]; };

// ---------------- WMMA fragment loaders (wave32, CDNA5 layouts) ----------------

// A (16x32 bf16, MxK): lane m=lane&15, half=lane>>4.
// elems 0..7  = K  8*half .. 8*half+7   (contiguous)
// elems 8..15 = K 16+8*half .. +7       (contiguous)
__device__ __forceinline__ v16bf a_frag_bf16(const __bf16* A, int ld, int kbase, int lane) {
    const int m = lane & 15, half = lane >> 4;
    const __bf16* p = A + m * ld + kbase + 8 * half;
    BF16x16 r;
    r.u[0] = *(const uint4*)(p);
    r.u[1] = *(const uint4*)(p + 16);
    return r.v;
}

// A fragment directly from an f32 row (embedding gather), converted in-register.
__device__ __forceinline__ v16bf a_frag_f32(const float* row, int kbase, int lane) {
    const int half = lane >> 4;
    const float* p = row + kbase + 8 * half;
    float4 f0 = *(const float4*)(p);
    float4 f1 = *(const float4*)(p + 4);
    float4 f2 = *(const float4*)(p + 16);
    float4 f3 = *(const float4*)(p + 20);
    BF16x16 r;
    r.v[0]  = (__bf16)f0.x; r.v[1]  = (__bf16)f0.y; r.v[2]  = (__bf16)f0.z; r.v[3]  = (__bf16)f0.w;
    r.v[4]  = (__bf16)f1.x; r.v[5]  = (__bf16)f1.y; r.v[6]  = (__bf16)f1.z; r.v[7]  = (__bf16)f1.w;
    r.v[8]  = (__bf16)f2.x; r.v[9]  = (__bf16)f2.y; r.v[10] = (__bf16)f2.z; r.v[11] = (__bf16)f2.w;
    r.v[12] = (__bf16)f3.x; r.v[13] = (__bf16)f3.y; r.v[14] = (__bf16)f3.z; r.v[15] = (__bf16)f3.w;
    return r.v;
}

// B (32x16 bf16, KxN) where B[k][n] = W[n][k], W row-major [N_total x K]:
// lane n=lane&15, half=lane>>4; elems 0..7 = K 8*half..+7, elems 8..15 = K 16+8*half..+7 of W row (col_base+n).
__device__ __forceinline__ v16bf b_frag_w(const __bf16* W, int ld, int col_base, int kbase, int lane) {
    const int n = lane & 15, half = lane >> 4;
    const __bf16* p = W + (size_t)(col_base + n) * ld + kbase + 8 * half;
    BF16x16 r;
    r.u[0] = *(const uint4*)(p);
    r.u[1] = *(const uint4*)(p + 16);
    return r.v;
}

__device__ __forceinline__ v8f wmma_acc(v16bf a, v16bf b, v8f c) {
    return __builtin_amdgcn_wmma_f32_16x16x32_bf16(false, a, false, b, (short)0, c, false, false);
}

// ---------------- grid-wide generation barrier ----------------

__device__ __forceinline__ void grid_barrier(int* counter, int* gen) {
    __syncthreads();
    if (threadIdx.x == 0) {
        __builtin_amdgcn_fence(__ATOMIC_RELEASE, "agent");
        int g = __hip_atomic_load(gen, __ATOMIC_RELAXED, __HIP_MEMORY_SCOPE_AGENT);
        int a = __hip_atomic_fetch_add(counter, 1, __ATOMIC_RELAXED, __HIP_MEMORY_SCOPE_AGENT);
        if (a == GRID_ - 1) {
            __hip_atomic_store(counter, 0, __ATOMIC_RELAXED, __HIP_MEMORY_SCOPE_AGENT);
            __hip_atomic_fetch_add(gen, 1, __ATOMIC_RELEASE, __HIP_MEMORY_SCOPE_AGENT);
        } else {
            while (__hip_atomic_load(gen, __ATOMIC_RELAXED, __HIP_MEMORY_SCOPE_AGENT) == g) {
                __builtin_amdgcn_s_sleep(1);
            }
        }
        __builtin_amdgcn_fence(__ATOMIC_ACQUIRE, "agent");
    }
    __syncthreads();
}

__device__ __forceinline__ float sigm(float x) { return 1.0f / (1.0f + __expf(-x)); }

// ---------------- setup kernels ----------------

__global__ void cvt_bf16_kernel(const float* __restrict__ src, __bf16* __restrict__ dst, int n) {
    int stride = gridDim.x * blockDim.x;
    for (int i = blockIdx.x * blockDim.x + threadIdx.x; i < n; i += stride)
        dst[i] = (__bf16)src[i];
}

__global__ void add_bias_kernel(const float* __restrict__ a, const float* __restrict__ b,
                                float* __restrict__ dst, int n) {
    int i = blockIdx.x * blockDim.x + threadIdx.x;
    if (i < n) dst[i] = a[i] + b[i];
}

// h0 = initial_state @ W_enc^T + b_enc ; broadcast into (h,c) of both layers.
__global__ void init_state_kernel(const float* __restrict__ init_state,
                                  const float* __restrict__ Wenc,
                                  const float* __restrict__ benc,
                                  float* c0s, float* c1s, __bf16* h0b, __bf16* h1b, int* bar) {
    int idx = blockIdx.x * blockDim.x + threadIdx.x;
    if (idx == 0) { bar[0] = 0; bar[1] = 0; }
    if (idx < B_ * H_) {
        int b = idx >> 9, j = idx & (H_ - 1);
        const float* x = init_state + (size_t)b * EH_;
        const float* w = Wenc + (size_t)j * EH_;
        float s = benc[j];
        #pragma unroll 4
        for (int k = 0; k < EH_; ++k) s += x[k] * w[k];
        c0s[idx] = s; c1s[idx] = s;
        h0b[idx] = (__bf16)s; h1b[idx] = (__bf16)s;
    }
}

// ---------------- fused per-timestep kernel ----------------

__global__ __launch_bounds__(TPB_) void lstm_step_kernel(
    const int*   __restrict__ input_ids,   // [B, T]
    const float* __restrict__ emb,         // [V, E] f32
    const __bf16* __restrict__ Wih0, const __bf16* __restrict__ Whh0,   // [2048,512]
    const __bf16* __restrict__ Wih1, const __bf16* __restrict__ Whh1,   // [2048,512]
    const __bf16* __restrict__ Wp,                                      // [32000,512]
    const float* __restrict__ b0, const float* __restrict__ b1,         // [2048] combined biases
    const float* __restrict__ bp,                                       // [32000]
    float* c0s, float* c1s, __bf16* h0b, __bf16* h1b,                   // state
    float* gates, int* bar, float* out, int t)
{
    const int lane = threadIdx.x & 31;
    const int wv   = blockIdx.x * (TPB_ / 32) + (threadIdx.x >> 5);  // global wave id, 0..1023
    const int n    = lane & 15;
    const int half = lane >> 4;

    // ---- Phase 1: layer-0 gates  g0 = x@Wih0^T + h0@Whh0^T + (b_ih0+b_hh0)
    if (wv < G4H_ / 16) {
        const int col = wv * 16;
        const int tok = input_ids[n * T_ + t];       // row m = lane&15 of the batch
        const float* xrow = emb + (size_t)tok * E_;  // embedding gather, fused
        v8f acc = {};
        #pragma unroll
        for (int kc = 0; kc < E_ / 32; ++kc) {
            v16bf a = a_frag_f32(xrow, kc * 32, lane);
            v16bf b = b_frag_w(Wih0, E_, col, kc * 32, lane);
            acc = wmma_acc(a, b, acc);
        }
        #pragma unroll
        for (int kc = 0; kc < H_ / 32; ++kc) {
            v16bf a = a_frag_bf16(h0b, H_, kc * 32, lane);
            v16bf b = b_frag_w(Whh0, H_, col, kc * 32, lane);
            acc = wmma_acc(a, b, acc);
        }
        const float bias = b0[col + n];
        #pragma unroll
        for (int r = 0; r < 8; ++r)
            gates[(r + 8 * half) * G4H_ + col + n] = acc[r] + bias;
    }
    grid_barrier(bar, bar + 1);

    // ---- Phase 2: layer-0 pointwise LSTM
    {
        int idx = blockIdx.x * TPB_ + threadIdx.x;
        if (idx < B_ * H_) {
            int b = idx >> 9, j = idx & (H_ - 1);
            const float* g = gates + b * G4H_;
            float gi = g[j], gf = g[H_ + j], gg = g[2 * H_ + j], go = g[3 * H_ + j];
            float c  = c0s[idx];
            float c2 = sigm(gf) * c + sigm(gi) * tanhf(gg);
            float h2 = sigm(go) * tanhf(c2);
            c0s[idx] = c2;
            h0b[idx] = (__bf16)h2;
        }
    }
    grid_barrier(bar, bar + 1);

    // ---- Phase 3: layer-1 gates  g1 = h0@Wih1^T + h1@Whh1^T + (b_ih1+b_hh1)
    if (wv < G4H_ / 16) {
        const int col = wv * 16;
        v8f acc = {};
        #pragma unroll
        for (int kc = 0; kc < H_ / 32; ++kc) {
            v16bf a = a_frag_bf16(h0b, H_, kc * 32, lane);
            v16bf b = b_frag_w(Wih1, H_, col, kc * 32, lane);
            acc = wmma_acc(a, b, acc);
        }
        #pragma unroll
        for (int kc = 0; kc < H_ / 32; ++kc) {
            v16bf a = a_frag_bf16(h1b, H_, kc * 32, lane);
            v16bf b = b_frag_w(Whh1, H_, col, kc * 32, lane);
            acc = wmma_acc(a, b, acc);
        }
        const float bias = b1[col + n];
        #pragma unroll
        for (int r = 0; r < 8; ++r)
            gates[(r + 8 * half) * G4H_ + col + n] = acc[r] + bias;
    }
    grid_barrier(bar, bar + 1);

    // ---- Phase 4: layer-1 pointwise LSTM
    {
        int idx = blockIdx.x * TPB_ + threadIdx.x;
        if (idx < B_ * H_) {
            int b = idx >> 9, j = idx & (H_ - 1);
            const float* g = gates + b * G4H_;
            float gi = g[j], gf = g[H_ + j], gg = g[2 * H_ + j], go = g[3 * H_ + j];
            float c  = c1s[idx];
            float c2 = sigm(gf) * c + sigm(gi) * tanhf(gg);
            float h2 = sigm(go) * tanhf(c2);
            c1s[idx] = c2;
            h1b[idx] = (__bf16)h2;
        }
    }
    grid_barrier(bar, bar + 1);

    // ---- Phase 5: vocab projection  logits = h1 @ Wp^T + bp  -> out[b][t][:]
    {
        const bool two  = (wv < PROJ_TILES_ - NWAVES_);   // wv < 976 handles 2nd tile
        const int  colA = wv * 16;
        const int  colB = (wv + NWAVES_) * 16;
        v8f ca = {}, cb = {};
        #pragma unroll
        for (int kc = 0; kc < H_ / 32; ++kc) {
            v16bf a  = a_frag_bf16(h1b, H_, kc * 32, lane);
            if (kc + 1 < H_ / 32)   // warm L0 for next K-chunk (global_prefetch_b8)
                __builtin_prefetch(Wp + (size_t)(colA + n) * H_ + (kc + 1) * 32, 0, 1);
            v16bf bA = b_frag_w(Wp, H_, colA, kc * 32, lane);
            ca = wmma_acc(a, bA, ca);
            if (two) {
                v16bf bB = b_frag_w(Wp, H_, colB, kc * 32, lane);
                cb = wmma_acc(a, bB, cb);
            }
        }
        const float biasA = bp[colA + n];
        #pragma unroll
        for (int r = 0; r < 8; ++r) {
            int m = r + 8 * half;
            out[((size_t)m * STEPS_ + t) * V_ + colA + n] = ca[r] + biasA;
        }
        if (two) {
            const float biasB = bp[colB + n];
            #pragma unroll
            for (int r = 0; r < 8; ++r) {
                int m = r + 8 * half;
                out[((size_t)m * STEPS_ + t) * V_ + colB + n] = cb[r] + biasB;
            }
        }
    }
}

// ---------------- argmax over V for every (b, t) ----------------

__global__ void argmax_kernel(const float* __restrict__ logits, float* __restrict__ preds) {
    __shared__ float sv[256];
    __shared__ int   si[256];
    const int row = blockIdx.x;                  // b*255 + t
    const float* p = logits + (size_t)row * V_;
    float best = -__builtin_inff(); int bi = 0;
    for (int v = threadIdx.x; v < V_; v += 256) {
        float x = p[v];
        if (x > best || (x == best && v < bi)) { best = x; bi = v; }
    }
    sv[threadIdx.x] = best; si[threadIdx.x] = bi;
    __syncthreads();
    for (int s = 128; s > 0; s >>= 1) {
        if (threadIdx.x < s) {
            float ov = sv[threadIdx.x + s]; int oi = si[threadIdx.x + s];
            if (ov > sv[threadIdx.x] || (ov == sv[threadIdx.x] && oi < si[threadIdx.x])) {
                sv[threadIdx.x] = ov; si[threadIdx.x] = oi;
            }
        }
        __syncthreads();
    }
    if (threadIdx.x == 0) preds[row] = (float)si[0];
}

// ---------------- workspace layout (bytes, all 256-aligned) ----------------
#define OFF_BAR   0u
#define OFF_C0    256u
#define OFF_C1    (OFF_C0 + B_*H_*4u)
#define OFF_H0B   (OFF_C1 + B_*H_*4u)
#define OFF_H1B   (OFF_H0B + B_*H_*2u)
#define OFF_G     (OFF_H1B + B_*H_*2u)
#define OFF_B0    (OFF_G + B_*G4H_*4u)
#define OFF_B1    (OFF_B0 + G4H_*4u)
#define OFF_WIH0  (OFF_B1 + G4H_*4u)
#define OFF_WHH0  (OFF_WIH0 + (unsigned)G4H_*E_*2u)
#define OFF_WIH1  (OFF_WHH0 + (unsigned)G4H_*H_*2u)
#define OFF_WHH1  (OFF_WIH1 + (unsigned)G4H_*H_*2u)
#define OFF_WP    (OFF_WHH1 + (unsigned)G4H_*H_*2u)

extern "C" void kernel_launch(void* const* d_in, const int* in_sizes, int n_in,
                              void* d_out, int out_size, void* d_ws, size_t ws_size,
                              hipStream_t stream) {
    const int*   input_ids     = (const int*)  d_in[0];
    const float* initial_state = (const float*)d_in[1];
    const float* emb           = (const float*)d_in[2];
    const float* W_enc         = (const float*)d_in[3];
    const float* b_enc         = (const float*)d_in[4];
    const float* W_ih0         = (const float*)d_in[5];
    const float* W_hh0         = (const float*)d_in[6];
    const float* b_ih0         = (const float*)d_in[7];
    const float* b_hh0         = (const float*)d_in[8];
    const float* W_ih1         = (const float*)d_in[9];
    const float* W_hh1         = (const float*)d_in[10];
    const float* b_ih1         = (const float*)d_in[11];
    const float* b_hh1         = (const float*)d_in[12];
    const float* Wp            = (const float*)d_in[13];
    const float* bp            = (const float*)d_in[14];

    char* ws = (char*)d_ws;
    int*    bar   = (int*)   (ws + OFF_BAR);
    float*  c0s   = (float*) (ws + OFF_C0);
    float*  c1s   = (float*) (ws + OFF_C1);
    __bf16* h0b   = (__bf16*)(ws + OFF_H0B);
    __bf16* h1b   = (__bf16*)(ws + OFF_H1B);
    float*  gates = (float*) (ws + OFF_G);
    float*  b0    = (float*) (ws + OFF_B0);
    float*  b1    = (float*) (ws + OFF_B1);
    __bf16* wih0b = (__bf16*)(ws + OFF_WIH0);
    __bf16* whh0b = (__bf16*)(ws + OFF_WHH0);
    __bf16* wih1b = (__bf16*)(ws + OFF_WIH1);
    __bf16* whh1b = (__bf16*)(ws + OFF_WHH1);
    __bf16* wpb   = (__bf16*)(ws + OFF_WP);

    float* logits = (float*)d_out;                       // [B, 255, V]
    float* preds  = logits + (size_t)B_ * STEPS_ * V_;   // [B, 255]

    // Setup: bf16 weight conversion, combined biases, encoder-projected state.
    hipLaunchKernelGGL(cvt_bf16_kernel, dim3(256),  dim3(256), 0, stream, W_ih0, wih0b, G4H_ * E_);
    hipLaunchKernelGGL(cvt_bf16_kernel, dim3(256),  dim3(256), 0, stream, W_hh0, whh0b, G4H_ * H_);
    hipLaunchKernelGGL(cvt_bf16_kernel, dim3(256),  dim3(256), 0, stream, W_ih1, wih1b, G4H_ * H_);
    hipLaunchKernelGGL(cvt_bf16_kernel, dim3(256),  dim3(256), 0, stream, W_hh1, whh1b, G4H_ * H_);
    hipLaunchKernelGGL(cvt_bf16_kernel, dim3(2048), dim3(256), 0, stream, Wp,    wpb,   V_ * H_);
    hipLaunchKernelGGL(add_bias_kernel, dim3(8),    dim3(256), 0, stream, b_ih0, b_hh0, b0, G4H_);
    hipLaunchKernelGGL(add_bias_kernel, dim3(8),    dim3(256), 0, stream, b_ih1, b_hh1, b1, G4H_);
    hipLaunchKernelGGL(init_state_kernel, dim3(32), dim3(256), 0, stream,
                       initial_state, W_enc, b_enc, c0s, c1s, h0b, h1b, bar);

    // Sequential recurrence: one fused kernel per timestep.
    for (int t = 0; t < STEPS_; ++t) {
        hipLaunchKernelGGL(lstm_step_kernel, dim3(GRID_), dim3(TPB_), 0, stream,
                           input_ids, emb, wih0b, whh0b, wih1b, whh1b, wpb,
                           b0, b1, bp, c0s, c1s, h0b, h1b, gates, bar, logits, t);
    }

    // Final argmax over V for each (b, t).
    hipLaunchKernelGGL(argmax_kernel, dim3(B_ * STEPS_), dim3(256), 0, stream, logits, preds);
}
